// Decoder_46342697123903
// MI455X (gfx1250) — compile-verified, matrix operand order checked by reading, not run on hardware
//
#include <hip/hip_runtime.h>
#include <hip/hip_bf16.h>
#include <math.h>

// ---------------- problem constants ----------------
#define NB   64      // batch
#define TT   64      // decode steps
#define SS   2048    // encoder length
#define DD   512     // hidden
#define DKQ  128     // key/query dim
#define VOC  32000
#define GAT  2048    // 4*D

typedef __attribute__((ext_vector_type(16))) __bf16 v16bf;
typedef __attribute__((ext_vector_type(8)))  __bf16 v8bf;
typedef __attribute__((ext_vector_type(8)))  float  v8f;

union BFrag { v16bf v; v8bf h2[2]; };

// ---------------- LDS staging helpers (async-to-LDS if available) ----------------
typedef int v4i_gcc __attribute__((vector_size(16)));   // matches builtin param pointee

__device__ __forceinline__ void stage_chunk16(__bf16* dst_lds, const __bf16* src_g) {
#if __has_builtin(__builtin_amdgcn_global_load_async_to_lds_b128)
    __builtin_amdgcn_global_load_async_to_lds_b128(
        (__attribute__((address_space(1))) v4i_gcc*)src_g,
        (__attribute__((address_space(3))) v4i_gcc*)dst_lds, 0, 0);
#else
    *(uint4*)dst_lds = *(const uint4*)src_g;
#endif
}

__device__ __forceinline__ void stage_wait_and_barrier() {
#if __has_builtin(__builtin_amdgcn_global_load_async_to_lds_b128)
#if __has_builtin(__builtin_amdgcn_s_wait_asynccnt)
    __builtin_amdgcn_s_wait_asynccnt(0);
#else
    asm volatile("s_wait_asynccnt 0x0" ::: "memory");
#endif
#endif
    __syncthreads();
}

// ---------------- f32 -> bf16 weight conversion ----------------
__global__ void f32_to_bf16_kernel(const float* __restrict__ src, __bf16* __restrict__ dst, int n) {
    int i = blockIdx.x * blockDim.x + threadIdx.x;
    int stride = gridDim.x * blockDim.x;
    for (; i < n; i += stride) dst[i] = (__bf16)src[i];
}

// ---------------- generic WMMA GEMM: OUT[64,N] = A1*W1^T (+ A2*W2^T) + bias1 (+bias2) ----------------
// A row-major [64, K], W row-major [N, K], K multiple of 32, N multiple of 64.
// Activations staged in LDS once per block (shared by all 4 waves) via async-to-LDS;
// weights stream from global/L2 with prefetch. One wave per 16-wide N strip; 4 M-tiles.
__global__ __launch_bounds__(128) void gemm_wmma_bf16(
        const __bf16* __restrict__ A1, const __bf16* __restrict__ W1, int K1,
        const __bf16* __restrict__ A2, const __bf16* __restrict__ W2, int K2,
        const float* __restrict__ bias1, const float* __restrict__ bias2,
        float* __restrict__ OUT, int N) {
    extern __shared__ __align__(16) __bf16 xs[];   // [64,K1] then [64,K2]
    const int tid  = threadIdx.x;
    const int wave = tid >> 5;
    const int lane = tid & 31;
    const int n0   = (blockIdx.x * 4 + wave) * 16;
    const int row  = lane & 15;           // M row inside tile for A, N row for W
    const int koff = (lane >> 4) * 8;     // 16-bit A/B layout: lanes 16-31 hold K+8 / K+24
    const int nn   = n0 + row;

    // ---- stage activations to LDS (16B chunks, flat copy) ----
    for (int i = tid; i < 8 * K1; i += 128)                      // 64*K1 elems / 8-per-chunk
        stage_chunk16(xs + (size_t)i * 8, A1 + (size_t)i * 8);
    if (A2) {
        __bf16* xs2 = xs + (size_t)64 * K1;
        for (int i = tid; i < 8 * K2; i += 128)
            stage_chunk16(xs2 + (size_t)i * 8, A2 + (size_t)i * 8);
    }
    stage_wait_and_barrier();

    float binit = 0.0f;
    if (bias1) binit += bias1[nn];
    if (bias2) binit += bias2[nn];
    v8f acc[4];
    #pragma unroll
    for (int mt = 0; mt < 4; ++mt)
        #pragma unroll
        for (int r = 0; r < 8; ++r) acc[mt][r] = binit;

    // ---- K loop over (LDS A1, global W1) ----
    {
        const __bf16* wrow = W1 + (size_t)nn * K1;
        for (int k0 = 0; k0 < K1; k0 += 32) {
            if (k0 + 32 < K1) __builtin_prefetch(wrow + k0 + 32, 0, 1); // global_prefetch_b8
            BFrag b;
            b.h2[0] = *(const v8bf*)(wrow + k0 + koff);
            b.h2[1] = *(const v8bf*)(wrow + k0 + koff + 16);
            // preload all four A fragments so DS loads clause together and
            // waits stagger under WMMA execution
            BFrag a[4];
            #pragma unroll
            for (int mt = 0; mt < 4; ++mt) {
                const __bf16* arow = xs + (size_t)(mt * 16 + row) * K1;
                a[mt].h2[0] = *(const v8bf*)(arow + k0 + koff);    // ds_load_b128
                a[mt].h2[1] = *(const v8bf*)(arow + k0 + koff + 16);
            }
            #pragma unroll
            for (int mt = 0; mt < 4; ++mt)
                acc[mt] = __builtin_amdgcn_wmma_f32_16x16x32_bf16(
                    false, a[mt].v, false, b.v, (short)0, acc[mt], false, false);
        }
    }
    // ---- optional second operand pair (fused  x*Wih^T + h*Whh^T) ----
    if (A2) {
        const __bf16* a2s  = xs + (size_t)64 * K1;
        const __bf16* wrow = W2 + (size_t)nn * K2;
        for (int k0 = 0; k0 < K2; k0 += 32) {
            if (k0 + 32 < K2) __builtin_prefetch(wrow + k0 + 32, 0, 1);
            BFrag b;
            b.h2[0] = *(const v8bf*)(wrow + k0 + koff);
            b.h2[1] = *(const v8bf*)(wrow + k0 + koff + 16);
            BFrag a[4];
            #pragma unroll
            for (int mt = 0; mt < 4; ++mt) {
                const __bf16* arow = a2s + (size_t)(mt * 16 + row) * K2;
                a[mt].h2[0] = *(const v8bf*)(arow + k0 + koff);
                a[mt].h2[1] = *(const v8bf*)(arow + k0 + koff + 16);
            }
            #pragma unroll
            for (int mt = 0; mt < 4; ++mt)
                acc[mt] = __builtin_amdgcn_wmma_f32_16x16x32_bf16(
                    false, a[mt].v, false, b.v, (short)0, acc[mt], false, false);
        }
    }
    // ---- store D: lane<16 -> M=r, lane>=16 -> M=r+8 ----
    const int mb = (lane >> 4) * 8;
    #pragma unroll
    for (int mt = 0; mt < 4; ++mt)
        #pragma unroll
        for (int r = 0; r < 8; ++r)
            OUT[(size_t)(mt * 16 + mb + r) * N + nn] = acc[mt][r];
}

// ---------------- x = [embedding[tok] | ctxt] -> bf16 [64,640] ----------------
__global__ void embed_concat_kernel(const int* __restrict__ input_txt,
                                    const float* __restrict__ embedding,
                                    const float* __restrict__ ctxt,
                                    __bf16* __restrict__ x_bf, int t) {
    int idx = blockIdx.x * blockDim.x + threadIdx.x;
    if (idx >= NB * (DD + DKQ)) return;
    int n = idx / (DD + DKQ), j = idx - n * (DD + DKQ);
    float v;
    if (j < DD) {
        int tok = input_txt[n * TT + t];
        v = embedding[(size_t)tok * DD + j];
    } else {
        v = ctxt[n * DKQ + (j - DD)];
    }
    x_bf[idx] = (__bf16)v;
}

// ---------------- y = [h3 | ctxt] -> bf16 [64,640] ----------------
__global__ void concat_h_ctxt_kernel(const float* __restrict__ h3,
                                     const float* __restrict__ ctxt,
                                     __bf16* __restrict__ y_bf) {
    int idx = blockIdx.x * blockDim.x + threadIdx.x;
    if (idx >= NB * (DD + DKQ)) return;
    int n = idx / (DD + DKQ), j = idx - n * (DD + DKQ);
    float v = (j < DD) ? h3[n * DD + j] : ctxt[n * DKQ + (j - DD)];
    y_bf[idx] = (__bf16)v;
}

// ---------------- LSTM pointwise: gates[64,2048] + c -> h,c + h_bf16 ----------------
__global__ void lstm_pointwise_kernel(const float* __restrict__ gates,
                                      float* __restrict__ h, float* __restrict__ c,
                                      __bf16* __restrict__ h_bf) {
    int idx = blockIdx.x * blockDim.x + threadIdx.x;
    if (idx >= NB * DD) return;
    int n = idx >> 9, j = idx & (DD - 1);
    const float* g = gates + (size_t)n * GAT;
    float gi = g[j], gf = g[DD + j], gg = g[2 * DD + j], go = g[3 * DD + j];
    float i_ = 1.0f / (1.0f + __expf(-gi));
    float f_ = 1.0f / (1.0f + __expf(-gf));
    float g_ = tanhf(gg);
    float o_ = 1.0f / (1.0f + __expf(-go));
    float c2 = f_ * c[idx] + i_ * g_;
    float h2v = o_ * tanhf(c2);
    c[idx] = c2; h[idx] = h2v; h_bf[idx] = (__bf16)h2v;
}

// ---------------- init state from h0/c0 broadcast ----------------
__global__ void init_state_kernel(const float* h01, const float* c01,
                                  const float* h02, const float* c02,
                                  const float* h03, const float* c03,
                                  float* h1, float* c1, float* h2, float* c2,
                                  float* h3, float* c3,
                                  __bf16* h1b, __bf16* h2b, __bf16* h3b) {
    int idx = blockIdx.x * blockDim.x + threadIdx.x;
    if (idx >= NB * DD) return;
    int j = idx & (DD - 1);
    h1[idx] = h01[j]; c1[idx] = c01[j]; h1b[idx] = (__bf16)h01[j];
    h2[idx] = h02[j]; c2[idx] = c02[j]; h2b[idx] = (__bf16)h02[j];
    h3[idx] = h03[j]; c3[idx] = c03[j]; h3b[idx] = (__bf16)h03[j];
}

// ---------------- attention: energy -> masked softmax -> L1 norm -> ctxt ----------------
// one block per batch row; 256 threads; S=2048, DK=128
__global__ __launch_bounds__(256) void attention_kernel(
        const float* __restrict__ q, const float* __restrict__ keys,
        const float* __restrict__ values, const int* __restrict__ seq_lens,
        float* __restrict__ att_out, float* __restrict__ ctxt) {
    __shared__ __align__(16) float qs[DKQ];
    __shared__ float att[SS];
    __shared__ float red[256];
    const int n = blockIdx.x, tid = threadIdx.x;
    if (tid < DKQ) qs[tid] = q[n * DKQ + tid];
    __syncthreads();
    const int slen = seq_lens[n];
    const float* kb = keys + (size_t)n * SS * DKQ;

    float e[SS / 256];
    float lmax = -INFINITY;
    #pragma unroll
    for (int i = 0; i < SS / 256; ++i) {
        int s = tid + i * 256;
        const float4* kr = (const float4*)(kb + (size_t)s * DKQ);
        float acc = 0.0f;
        #pragma unroll 8
        for (int d4 = 0; d4 < DKQ / 4; ++d4) {
            float4 kv = kr[d4];
            float4 qv = *(const float4*)(qs + d4 * 4);
            acc += kv.x * qv.x + kv.y * qv.y + kv.z * qv.z + kv.w * qv.w;
        }
        e[i] = acc;
        lmax = fmaxf(lmax, acc);
    }
    red[tid] = lmax; __syncthreads();
    for (int st = 128; st > 0; st >>= 1) {
        if (tid < st) red[tid] = fmaxf(red[tid], red[tid + st]);
        __syncthreads();
    }
    const float gmax = red[0]; __syncthreads();

    float lsum_all = 0.0f, lsum_m = 0.0f;
    #pragma unroll
    for (int i = 0; i < SS / 256; ++i) {
        int s = tid + i * 256;
        float ex = __expf(e[i] - gmax);
        lsum_all += ex;
        float exm = (s < slen) ? ex : 0.0f;
        lsum_m += exm;
        att[s] = exm;
    }
    red[tid] = lsum_all; __syncthreads();
    for (int st = 128; st > 0; st >>= 1) { if (tid < st) red[tid] += red[tid + st]; __syncthreads(); }
    const float sum_all = red[0]; __syncthreads();
    red[tid] = lsum_m; __syncthreads();
    for (int st = 128; st > 0; st >>= 1) { if (tid < st) red[tid] += red[tid + st]; __syncthreads(); }
    const float sum_m = red[0]; __syncthreads();

    // reference: softmax(e)/sum_all, *mask, / max(sum_masked/sum_all, eps)
    const float sm_sum = sum_m / sum_all;
    const float inv = 1.0f / (sum_all * fmaxf(sm_sum, 1e-12f));
    #pragma unroll
    for (int i = 0; i < SS / 256; ++i) {
        int s = tid + i * 256;
        float a = att[s] * inv;
        att[s] = a;
        att_out[(size_t)n * SS + s] = a;
    }
    __syncthreads();

    // ctxt[n,d] = sum_s att[s] * values[n,s,d]; 2 threads per d (halves of S)
    const int d = tid & (DKQ - 1), half = tid >> 7;
    const float* vb = values + (size_t)n * SS * DKQ;
    float acc = 0.0f;
    const int sbeg = half * (SS / 2);
    for (int s = sbeg; s < sbeg + SS / 2; ++s)
        acc += att[s] * vb[(size_t)s * DKQ + d];
    red[tid] = acc; __syncthreads();
    if (half == 0) ctxt[n * DKQ + d] = red[tid] + red[tid + 128];
}

// ---------------- argmax over logits row [VOC] ----------------
__global__ __launch_bounds__(256) void argmax_kernel(const float* __restrict__ logits,
                                                     float* __restrict__ gen) {
    __shared__ float vmax[256];
    __shared__ int vidx[256];
    const int n = blockIdx.x, tid = threadIdx.x;
    const float* row = logits + (size_t)n * VOC;
    float bm = -INFINITY; int bi = 0;
    for (int j = tid; j < VOC; j += 256) {
        float v = row[j];
        if (v > bm) { bm = v; bi = j; }
    }
    vmax[tid] = bm; vidx[tid] = bi; __syncthreads();
    for (int st = 128; st > 0; st >>= 1) {
        if (tid < st) {
            float vo = vmax[tid + st]; int io = vidx[tid + st];
            if (vo > vmax[tid] || (vo == vmax[tid] && io < vidx[tid])) { vmax[tid] = vo; vidx[tid] = io; }
        }
        __syncthreads();
    }
    if (tid == 0) gen[n] = (float)vidx[0];
}

// ---------------- host side ----------------
extern "C" void kernel_launch(void* const* d_in, const int* in_sizes, int n_in,
                              void* d_out, int out_size, void* d_ws, size_t ws_size,
                              hipStream_t stream) {
    (void)in_sizes; (void)n_in; (void)out_size; (void)ws_size;
    // inputs in setup order
    const int*   input_txt = (const int*)  d_in[0];
    const float* keys      = (const float*)d_in[2];
    const float* values    = (const float*)d_in[3];
    const int*   seq_lens  = (const int*)  d_in[4];
    const float* embedding = (const float*)d_in[5];
    const float* W_ih[3] = { (const float*)d_in[6],  (const float*)d_in[12], (const float*)d_in[18] };
    const float* W_hh[3] = { (const float*)d_in[7],  (const float*)d_in[13], (const float*)d_in[19] };
    const float* b_ih[3] = { (const float*)d_in[8],  (const float*)d_in[14], (const float*)d_in[20] };
    const float* b_hh[3] = { (const float*)d_in[9],  (const float*)d_in[15], (const float*)d_in[21] };
    const float* h0[3]   = { (const float*)d_in[10], (const float*)d_in[16], (const float*)d_in[22] };
    const float* c0[3]   = { (const float*)d_in[11], (const float*)d_in[17], (const float*)d_in[23] };
    const float* Wq   = (const float*)d_in[24];
    const float* bq   = (const float*)d_in[25];
    const float* Wout = (const float*)d_in[26];
    const float* bout = (const float*)d_in[27];

    // output slices: logits [T,N,V], attentions [T,N,S], generateds [T,N]
    float* logits_base = (float*)d_out;
    float* att_base    = logits_base + (size_t)TT * NB * VOC;
    float* gen_base    = att_base    + (size_t)TT * NB * SS;

    // workspace carve-up (256B aligned)
    char* wsb = (char*)d_ws;
    size_t off = 0;
    auto carve = [&](size_t bytes) -> void* {
        off = (off + 255) & ~(size_t)255;
        void* p = wsb + off;
        off += bytes;
        return p;
    };
    const int KIN = DD + DKQ; // 640
    __bf16* Wout_bf = (__bf16*)carve((size_t)VOC * KIN * 2);
    __bf16* Wih_bf[3], *Whh_bf[3];
    Wih_bf[0] = (__bf16*)carve((size_t)GAT * KIN * 2);
    Whh_bf[0] = (__bf16*)carve((size_t)GAT * DD * 2);
    for (int l = 1; l < 3; ++l) {
        Wih_bf[l] = (__bf16*)carve((size_t)GAT * DD * 2);
        Whh_bf[l] = (__bf16*)carve((size_t)GAT * DD * 2);
    }
    __bf16* Wq_bf = (__bf16*)carve((size_t)DKQ * DD * 2);
    __bf16* x_bf  = (__bf16*)carve((size_t)NB * KIN * 2);
    __bf16* y_bf  = (__bf16*)carve((size_t)NB * KIN * 2);
    float* h1 = (float*)carve((size_t)NB * DD * 4);
    float* c1 = (float*)carve((size_t)NB * DD * 4);
    float* h2 = (float*)carve((size_t)NB * DD * 4);
    float* c2 = (float*)carve((size_t)NB * DD * 4);
    float* h3 = (float*)carve((size_t)NB * DD * 4);
    float* c3 = (float*)carve((size_t)NB * DD * 4);
    __bf16* h1b = (__bf16*)carve((size_t)NB * DD * 2);
    __bf16* h2b = (__bf16*)carve((size_t)NB * DD * 2);
    __bf16* h3b = (__bf16*)carve((size_t)NB * DD * 2);
    float* ctxt  = (float*)carve((size_t)NB * DKQ * 4);
    float* qbuf  = (float*)carve((size_t)NB * DKQ * 4);
    float* gates = (float*)carve((size_t)NB * GAT * 4);

    // allow the large dynamic LDS allocations (up to 144KB for the fused gate GEMM)
    (void)hipFuncSetAttribute((const void*)gemm_wmma_bf16,
                              hipFuncAttributeMaxDynamicSharedMemorySize, 160 * 1024);

    // dynamic LDS sizes per GEMM instance
    const size_t sh_gate1 = (size_t)(KIN + DD) * NB * 2;   // 144 KB
    const size_t sh_gate  = (size_t)(DD + DD) * NB * 2;    // 128 KB
    const size_t sh_q     = (size_t)DD * NB * 2;           //  64 KB
    const size_t sh_out   = (size_t)KIN * NB * 2;          //  80 KB

    // ---- one-time (per call) weight conversion f32 -> bf16 ----
    auto conv = [&](const float* s, __bf16* d, size_t n) {
        int blocks = (int)((n + 255) / 256); if (blocks > 2048) blocks = 2048;
        f32_to_bf16_kernel<<<blocks, 256, 0, stream>>>(s, d, (int)n);
    };
    conv(Wout,    Wout_bf,  (size_t)VOC * KIN);
    conv(W_ih[0], Wih_bf[0],(size_t)GAT * KIN);
    conv(W_hh[0], Whh_bf[0],(size_t)GAT * DD);
    conv(W_ih[1], Wih_bf[1],(size_t)GAT * DD);
    conv(W_hh[1], Whh_bf[1],(size_t)GAT * DD);
    conv(W_ih[2], Wih_bf[2],(size_t)GAT * DD);
    conv(W_hh[2], Whh_bf[2],(size_t)GAT * DD);
    conv(Wq,      Wq_bf,    (size_t)DKQ * DD);

    // ---- init state + initial context (q0 = h3_init @ Wq^T + bq) ----
    init_state_kernel<<<(NB * DD + 255) / 256, 256, 0, stream>>>(
        h0[0], c0[0], h0[1], c0[1], h0[2], c0[2],
        h1, c1, h2, c2, h3, c3, h1b, h2b, h3b);
    gemm_wmma_bf16<<<DKQ / 64, 128, sh_q, stream>>>(
        h3b, Wq_bf, DD, nullptr, nullptr, 0, bq, nullptr, qbuf, DKQ);
    attention_kernel<<<NB, 256, 0, stream>>>(qbuf, keys, values, seq_lens,
                                             att_base /*scratch write, overwritten at t=0*/, ctxt);

    const int ccat_blocks = (NB * KIN + 255) / 256;
    const int pw_blocks   = (NB * DD + 255) / 256;

    for (int t = 0; t < TT; ++t) {
        float* logits_t = logits_base + (size_t)t * NB * VOC;
        float* att_t    = att_base    + (size_t)t * NB * SS;
        float* gen_t    = gen_base    + (size_t)t * NB;

        embed_concat_kernel<<<ccat_blocks, 256, 0, stream>>>(input_txt, embedding, ctxt, x_bf, t);

        // cell 1: gates = x@Wih1^T + h1@Whh1^T + b
        gemm_wmma_bf16<<<GAT / 64, 128, sh_gate1, stream>>>(
            x_bf, Wih_bf[0], KIN, h1b, Whh_bf[0], DD, b_ih[0], b_hh[0], gates, GAT);
        lstm_pointwise_kernel<<<pw_blocks, 256, 0, stream>>>(gates, h1, c1, h1b);
        // cell 2
        gemm_wmma_bf16<<<GAT / 64, 128, sh_gate, stream>>>(
            h1b, Wih_bf[1], DD, h2b, Whh_bf[1], DD, b_ih[1], b_hh[1], gates, GAT);
        lstm_pointwise_kernel<<<pw_blocks, 256, 0, stream>>>(gates, h2, c2, h2b);
        // cell 3
        gemm_wmma_bf16<<<GAT / 64, 128, sh_gate, stream>>>(
            h2b, Wih_bf[2], DD, h3b, Whh_bf[2], DD, b_ih[2], b_hh[2], gates, GAT);
        lstm_pointwise_kernel<<<pw_blocks, 256, 0, stream>>>(gates, h3, c3, h3b);

        // q = h3 @ Wq^T + bq ; attention ; ctxt
        gemm_wmma_bf16<<<DKQ / 64, 128, sh_q, stream>>>(
            h3b, Wq_bf, DD, nullptr, nullptr, 0, bq, nullptr, qbuf, DKQ);
        attention_kernel<<<NB, 256, 0, stream>>>(qbuf, keys, values, seq_lens, att_t, ctxt);

        // out = [h3|ctxt] @ Wout^T + bout -> logits ; argmax
        concat_h_ctxt_kernel<<<ccat_blocks, 256, 0, stream>>>(h3, ctxt, y_bf);
        gemm_wmma_bf16<<<VOC / 64, 128, sh_out, stream>>>(
            y_bf, Wout_bf, KIN, nullptr, nullptr, 0, bout, nullptr, logits_t, VOC);
        argmax_kernel<<<NB, 256, 0, stream>>>(logits_t, gen_t);
    }
}
